// MultiHeadAttention_39694087750294
// MI455X (gfx1250) — matrix operand order repaired and checked by reading, not compiled, not run on hardware
//
#include <hip/hip_runtime.h>

typedef __attribute__((ext_vector_type(16))) __bf16 v16bf;
typedef __attribute__((ext_vector_type(8)))  float  v8f;
typedef int v4i __attribute__((vector_size(16)));

constexpr int Bc = 4;       // batch
constexpr int Sc = 2048;    // sequence
constexpr int Ec = 1024;    // embed
constexpr int Hc = 8;       // heads
constexpr int Dc = 128;     // head dim
constexpr int Mc = Bc * Sc; // 8192 rows

#if defined(__HIP_DEVICE_COMPILE__) &&                                   \
    __has_builtin(__builtin_amdgcn_global_load_async_to_lds_b128) &&     \
    __has_builtin(__builtin_amdgcn_s_wait_asynccnt)
#define USE_ASYNC 1
#define GPTR(p) ((__attribute__((address_space(1))) v4i*)(p))
#define LPTR(p) ((__attribute__((address_space(3))) v4i*)(p))
#else
#define USE_ASYNC 0
#endif

__device__ __forceinline__ v8f wmma_bf16(v16bf a, v16bf b, v8f c) {
  return __builtin_amdgcn_wmma_f32_16x16x32_bf16(
      /*neg_a=*/false, a, /*neg_b=*/false, b,
      /*c_mod=*/(short)0, c, /*reuse_a=*/false, /*reuse_b=*/false);
}

// ---------------------------------------------------------------- convert
__global__ __launch_bounds__(256) void f32_to_bf16_kernel(
    const float* __restrict__ in, __bf16* __restrict__ out, int n) {
  int i = blockIdx.x * blockDim.x + threadIdx.x;
  int stride = gridDim.x * blockDim.x;
  for (; i < n; i += stride) out[i] = (__bf16)in[i];
}

// ---------------------------------------------------------------- GEMM
// C[M,N] (+z offsets) = A[M,K] * B[K,N] (both row major), bf16 in, f32
// accumulate, store f32 (Cf) or bf16 (Cb). 128x128 tile per block, 8 waves,
// each wave: 16 rows x 128 cols (8 WMMA accum tiles per K-step of 32).
__global__ __launch_bounds__(256) void gemm_bf16_kernel(
    const __bf16* __restrict__ A, const __bf16* __restrict__ Bm,
    float* __restrict__ Cf, __bf16* __restrict__ Cb,
    int K, int N, int ldc, long aZ, long bZ, long cZ) {
  __shared__ __bf16 As[128][32 + 8];   // [m][k]
  __shared__ __bf16 Bs[128][32 + 8];   // transposed: [n][k]

  const int tid = threadIdx.x, lane = tid & 31, wave = tid >> 5;
  const int z = blockIdx.z;
  const __bf16* Ab = A + (long)blockIdx.x * 128 * K + aZ * z;
  const __bf16* Bb = Bm + (long)blockIdx.y * 128 + bZ * z;
  const long cbase = (long)blockIdx.x * 128 * ldc + (long)blockIdx.y * 128 + cZ * z;

  v8f zero = {};
  v8f acc[8];
#pragma unroll
  for (int t = 0; t < 8; ++t) acc[t] = zero;

  for (int k0 = 0; k0 < K; k0 += 32) {
    {  // A tile: 128x32, each thread 32 contiguous bytes (async to LDS)
      int r = tid >> 1, c = (tid & 1) * 16;
      const __bf16* src = Ab + (long)r * K + k0 + c;
#if USE_ASYNC
      __builtin_amdgcn_global_load_async_to_lds_b128(GPTR(src), LPTR(&As[r][c]), 0, 0);
      __builtin_amdgcn_global_load_async_to_lds_b128(GPTR(src + 8), LPTR(&As[r][c + 8]), 0, 0);
#else
#pragma unroll
      for (int i = 0; i < 16; ++i) As[r][c + i] = src[i];
#endif
    }
    {  // B tile: 32x128 from global, stored transposed [n][k]
      int r = tid >> 3, c = (tid & 7) * 16;
      const __bf16* src = Bb + (long)(k0 + r) * N + c;
#pragma unroll
      for (int i = 0; i < 16; ++i) Bs[c + i][r] = src[i];
    }
#if USE_ASYNC
    __builtin_amdgcn_s_wait_asynccnt(0);
#endif
    __syncthreads();

    const int m0 = wave * 16;
    v16bf af;
    {  // A fragment: ISA 16-bit A layout (16x32)
      int m = m0 + (lane & 15);
      int kb = (lane >> 4) * 8;
#pragma unroll
      for (int r = 0; r < 8; ++r) {
        int kk = kb + ((r & 4) << 2) + ((r & 3) << 1);
        af[2 * r] = As[m][kk];
        af[2 * r + 1] = As[m][kk + 1];
      }
    }
    v16bf bfr[8];  // preload all B fragments, then back-to-back WMMAs
#pragma unroll
    for (int t = 0; t < 8; ++t) {
      int n = t * 16 + (lane & 15);
      int kb = (lane >> 4) * 8;
      v16bf f;
#pragma unroll
      for (int r = 0; r < 8; ++r) {
        int kk = kb + ((r & 4) << 2) + ((r & 3) << 1);
        f[2 * r] = Bs[n][kk];
        f[2 * r + 1] = Bs[n][kk + 1];
      }
      bfr[t] = f;
    }
#pragma unroll
    for (int t = 0; t < 8; ++t) acc[t] = wmma_bf16(af, bfr[t], acc[t]);
    __syncthreads();
  }

  // store: C/D layout — lane<16: (M=r, N=lane); lane>=16: (M=8+r, N=lane-16)
  const int m0 = wave * 16;
  const int nL = lane & 15;
  const int mb = m0 + ((lane >> 4) << 3);
  if (Cf) {
#pragma unroll
    for (int t = 0; t < 8; ++t)
#pragma unroll
      for (int r = 0; r < 8; ++r)
        Cf[cbase + (long)(mb + r) * ldc + t * 16 + nL] = acc[t][r];
  } else {
#pragma unroll
    for (int t = 0; t < 8; ++t)
#pragma unroll
      for (int r = 0; r < 8; ++r)
        Cb[cbase + (long)(mb + r) * ldc + t * 16 + nL] = (__bf16)acc[t][r];
  }
}

// ---------------------------------------------------------------- attention
// One block: 128 query rows of one (b, h). 8 waves x 16 rows. Flash-style
// streaming softmax over 32-key steps. Q/K/V in (h,b,s,d) bf16; O written
// (b,s,h,d) bf16 for the final projection GEMM.
__global__ __launch_bounds__(256) void attn_kernel(
    const __bf16* __restrict__ Q, const __bf16* __restrict__ Kt,
    const __bf16* __restrict__ V, const int* __restrict__ mask,
    __bf16* __restrict__ O) {
  __shared__ __bf16 Ks[32][128 + 8];     // [key][d]
  __shared__ __bf16 Vs[128][32 + 8];     // transposed: [d][key]
  __shared__ __bf16 Ps[8][16][32 + 2];   // per-wave P scratch
  __shared__ unsigned char Ms[128][36];  // mask tile [qrow][key]

  const int tid = threadIdx.x, lane = tid & 31, wave = tid >> 5;
  const int s0 = blockIdx.x * 128;
  const int b = blockIdx.y, h = blockIdx.z;

  const long base = ((long)h * Bc + b) * Sc * Dc;
  const __bf16* Qh = Q + base;
  const __bf16* Kh = Kt + base;
  const __bf16* Vh = V + base;
  const int* mrow = mask + (long)b * Sc * Sc;

  const int myq = s0 + wave * 16;
  const int nL = lane & 15;
  const int mb = (lane >> 4) << 3;

  // Q fragments: 4 chunks of 32 over D=128, held in registers
  v16bf qf[4];
  {
    int m = myq + nL;
    int kb = (lane >> 4) * 8;
#pragma unroll
    for (int j = 0; j < 4; ++j)
#pragma unroll
      for (int r = 0; r < 8; ++r) {
        int kk = j * 32 + kb + ((r & 4) << 2) + ((r & 3) << 1);
        qf[j][2 * r] = Qh[(long)m * Dc + kk];
        qf[j][2 * r + 1] = Qh[(long)m * Dc + kk + 1];
      }
  }

  v8f zero = {};
  v8f acc[8];
#pragma unroll
  for (int t = 0; t < 8; ++t) acc[t] = zero;
  float mI[8], lI[8];
#pragma unroll
  for (int r = 0; r < 8; ++r) { mI[r] = -1e30f; lI[r] = 0.f; }

  const float scale = 0.08838834764831845f;  // 1/sqrt(128)

  for (int kb0 = 0; kb0 < Sc; kb0 += 32) {
    {  // K tile (async, row-contiguous), V tile transposed, mask tile bytes
      int r = tid >> 3, c = (tid & 7) * 16;
      const __bf16* ksrc = Kh + (long)(kb0 + r) * Dc + c;
#if USE_ASYNC
      __builtin_amdgcn_global_load_async_to_lds_b128(GPTR(ksrc), LPTR(&Ks[r][c]), 0, 0);
      __builtin_amdgcn_global_load_async_to_lds_b128(GPTR(ksrc + 8), LPTR(&Ks[r][c + 8]), 0, 0);
#else
#pragma unroll
      for (int i = 0; i < 16; ++i) Ks[r][c + i] = ksrc[i];
#endif
      const __bf16* vsrc = Vh + (long)(kb0 + r) * Dc + c;
#pragma unroll
      for (int i = 0; i < 16; ++i) Vs[c + i][r] = vsrc[i];
      int mr = tid >> 1, mc = (tid & 1) * 16;
      const int* msrc = mrow + (long)(s0 + mr) * Sc + kb0 + mc;
#pragma unroll
      for (int i = 0; i < 16; ++i) Ms[mr][mc + i] = (unsigned char)(msrc[i] != 0);
#if USE_ASYNC
      __builtin_amdgcn_s_wait_asynccnt(0);
#endif
    }
    __syncthreads();

    // scores: preload 8 K^T fragments, then interleave two accumulators
    v16bf kf[8];  // [u*4 + j]
#pragma unroll
    for (int u = 0; u < 2; ++u)
#pragma unroll
      for (int j = 0; j < 4; ++j) {
        int n = u * 16 + nL;
        int kb = j * 32 + (lane >> 4) * 8;
        v16bf f;
#pragma unroll
        for (int r = 0; r < 8; ++r) {
          int kk = kb + ((r & 4) << 2) + ((r & 3) << 1);
          f[2 * r] = Ks[n][kk];
          f[2 * r + 1] = Ks[n][kk + 1];
        }
        kf[u * 4 + j] = f;
      }
    v8f s0v = zero, s1v = zero;
#pragma unroll
    for (int j = 0; j < 4; ++j) {
      s0v = wmma_bf16(qf[j], kf[j], s0v);
      s1v = wmma_bf16(qf[j], kf[4 + j], s1v);
    }
    v8f sc[2];
#pragma unroll
    for (int r = 0; r < 8; ++r) {
      sc[0][r] = s0v[r] * scale;
      sc[1][r] = s1v[r] * scale;
    }

    // mask (True = masked out), staged in LDS
#pragma unroll
    for (int u = 0; u < 2; ++u)
#pragma unroll
      for (int r = 0; r < 8; ++r)
        if (Ms[wave * 16 + mb + r][u * 16 + nL]) sc[u][r] = -1e30f;

    // online softmax: row reductions across the 16 lanes of each half-wave
    float corr[8];
#pragma unroll
    for (int r = 0; r < 8; ++r) {
      float x = fmaxf(sc[0][r], sc[1][r]);
#pragma unroll
      for (int off = 8; off >= 1; off >>= 1) x = fmaxf(x, __shfl_xor(x, off, 32));
      float mNew = fmaxf(mI[r], x);
      float p0 = (sc[0][r] <= -1e29f) ? 0.f : __expf(sc[0][r] - mNew);
      float p1 = (sc[1][r] <= -1e29f) ? 0.f : __expf(sc[1][r] - mNew);
      sc[0][r] = p0;
      sc[1][r] = p1;
      float rs = p0 + p1;
#pragma unroll
      for (int off = 8; off >= 1; off >>= 1) rs += __shfl_xor(rs, off, 32);
      corr[r] = __expf(mI[r] - mNew);
      lI[r] = lI[r] * corr[r] + rs;
      mI[r] = mNew;
    }
#pragma unroll
    for (int t = 0; t < 8; ++t)
#pragma unroll
      for (int r = 0; r < 8; ++r) acc[t][r] *= corr[r];

    // P (C-layout f32) -> per-wave LDS scratch -> A-layout bf16 fragment
#pragma unroll
    for (int u = 0; u < 2; ++u)
#pragma unroll
      for (int r = 0; r < 8; ++r)
        Ps[wave][mb + r][u * 16 + nL] = (__bf16)sc[u][r];
    v16bf pf;
    {
      int kb = (lane >> 4) * 8;
#pragma unroll
      for (int r = 0; r < 8; ++r) {
        int kk = kb + ((r & 4) << 2) + ((r & 3) << 1);
        pf[2 * r] = Ps[wave][nL][kk];
        pf[2 * r + 1] = Ps[wave][nL][kk + 1];
      }
    }
    // O += P * V: preload 8 V fragments (transposed Vs: pairs contiguous)
    v16bf vfr[8];
#pragma unroll
    for (int t = 0; t < 8; ++t) {
      int n = t * 16 + nL;
      int kb = (lane >> 4) * 8;
      v16bf f;
#pragma unroll
      for (int r = 0; r < 8; ++r) {
        int kk = kb + ((r & 4) << 2) + ((r & 3) << 1);
        f[2 * r] = Vs[n][kk];
        f[2 * r + 1] = Vs[n][kk + 1];
      }
      vfr[t] = f;
    }
#pragma unroll
    for (int t = 0; t < 8; ++t) acc[t] = wmma_bf16(pf, vfr[t], acc[t]);
    __syncthreads();
  }

  // normalize and store O in (b, s, h, d)
#pragma unroll
  for (int r = 0; r < 8; ++r) {
    float inv = (lI[r] > 0.f) ? 1.f / lI[r] : 0.f;
    long row = (long)(b * Sc + myq + mb + r) * (Hc * Dc) + h * Dc;
#pragma unroll
    for (int t = 0; t < 8; ++t)
      O[row + t * 16 + nL] = (__bf16)(acc[t][r] * inv);
  }
}

// ---------------------------------------------------------------- launch
extern "C" void kernel_launch(void* const* d_in, const int* in_sizes, int n_in,
                              void* d_out, int out_size, void* d_ws, size_t ws_size,
                              hipStream_t stream) {
  const float* q = (const float*)d_in[0];
  const int* mask = (const int*)d_in[1];
  const float* wq = (const float*)d_in[2];
  const float* wk = (const float*)d_in[3];
  const float* wv = (const float*)d_in[4];
  const float* wo = (const float*)d_in[5];
  float* out = (float*)d_out;

  char* ws = (char*)d_ws;
  auto take = [&](size_t nelem) {
    __bf16* p = (__bf16*)ws;
    ws += nelem * sizeof(__bf16);
    return p;
  };
  __bf16* qb = take((size_t)Mc * Ec);
  __bf16* wqb = take((size_t)Hc * Ec * Dc);
  __bf16* wkb = take((size_t)Hc * Ec * Dc);
  __bf16* wvb = take((size_t)Hc * Ec * Dc);
  __bf16* wob = take((size_t)Hc * Dc * Ec);
  __bf16* Qb = take((size_t)Hc * Mc * Dc);
  __bf16* Kb = take((size_t)Hc * Mc * Dc);
  __bf16* Vb = take((size_t)Hc * Mc * Dc);
  __bf16* Ob = take((size_t)Mc * Hc * Dc);

  auto cvt = [&](const float* src, __bf16* dst, int n) {
    int blocks = (n + 1023) / 1024;
    f32_to_bf16_kernel<<<blocks, 256, 0, stream>>>(src, dst, n);
  };
  cvt(q, qb, Mc * Ec);
  cvt(wq, wqb, Hc * Ec * Dc);
  cvt(wk, wkb, Hc * Ec * Dc);
  cvt(wv, wvb, Hc * Ec * Dc);
  cvt(wo, wob, Hc * Dc * Ec);

  // QKV projections: per head, C[h] (8192x128) = qb (8192x1024) * w[h] (1024x128)
  dim3 gqkv(Mc / 128, Dc / 128, Hc);
  gemm_bf16_kernel<<<gqkv, 256, 0, stream>>>(qb, wqb, nullptr, Qb, Ec, Dc, Dc,
                                             0, (long)Ec * Dc, (long)Mc * Dc);
  gemm_bf16_kernel<<<gqkv, 256, 0, stream>>>(qb, wkb, nullptr, Kb, Ec, Dc, Dc,
                                             0, (long)Ec * Dc, (long)Mc * Dc);
  gemm_bf16_kernel<<<gqkv, 256, 0, stream>>>(qb, wvb, nullptr, Vb, Ec, Dc, Dc,
                                             0, (long)Ec * Dc, (long)Mc * Dc);

  // flash attention
  dim3 ga(Sc / 128, Bc, Hc);
  attn_kernel<<<ga, 256, 0, stream>>>(Qb, Kb, Vb, mask, Ob);

  // output projection: out (8192x1024) = Ob (8192x1024) * w_out (1024x1024)
  dim3 go(Mc / 128, Ec / 128, 1);
  gemm_bf16_kernel<<<go, 256, 0, stream>>>(Ob, wob, out, nullptr, Hc * Dc, Ec, Ec,
                                           0, 0, 0);
}
